// _ViTAttn_18056042513119
// MI455X (gfx1250) — matrix operand order repaired and checked by reading, not compiled, not run on hardware
//
#include <hip/hip_runtime.h>

#define DIM   768
#define HEADS 12
#define HD    64
#define SEQ   2048
#define BATCH 4
#define MTOT  (BATCH*SEQ)   /* 8192 */
#define NQKV  (3*DIM)       /* 2304 */
#define ATTN_SCALE 0.125f   /* 1/sqrt(64) */

typedef __attribute__((ext_vector_type(16))) __bf16 v16bf;
typedef __attribute__((ext_vector_type(8)))  float  v8f;
typedef __attribute__((ext_vector_type(4)))  int    i32x4;
typedef unsigned int   u32x4 __attribute__((ext_vector_type(4)));
typedef float          f32x4 __attribute__((ext_vector_type(4)));
typedef unsigned short u16x4 __attribute__((ext_vector_type(4)));

#if defined(__has_builtin)
#if __has_builtin(__builtin_amdgcn_global_load_async_to_lds_b128) && \
    __has_builtin(__builtin_amdgcn_s_wait_asynccnt)
#define USE_ASYNC_LDS 1
#endif
#endif
#ifndef USE_ASYNC_LDS
#define USE_ASYNC_LDS 0
#endif

union FragBF {
  v16bf v;
  u32x4 q[2];
};

__device__ __forceinline__ unsigned short f2bf(float f) {
  unsigned u = __builtin_bit_cast(unsigned, f);
  u += 0x7FFFu + ((u >> 16) & 1u);            // round-to-nearest-even
  return (unsigned short)(u >> 16);
}

__device__ __forceinline__ v8f v8f_zero() {
  v8f z;
#pragma unroll
  for (int i = 0; i < 8; ++i) z[i] = 0.0f;
  return z;
}

__device__ __forceinline__ v8f wmma_bf16(v16bf a, v16bf b, v8f c) {
  // D = A(16x32 bf16) * B(32x16 bf16) + C(16x16 f32)
  return __builtin_amdgcn_wmma_f32_16x16x32_bf16(false, a, false, b, (short)0, c,
                                                 false, false);
}

// one 16-byte global -> LDS copy (async path if available)
__device__ __forceinline__ void cp_g2l_b128(const unsigned short* src,
                                            unsigned short* dst) {
#if USE_ASYNC_LDS
  __builtin_amdgcn_global_load_async_to_lds_b128(
      (__attribute__((address_space(1))) i32x4*)(unsigned long long)src,
      (__attribute__((address_space(3))) i32x4*)(unsigned)(unsigned long long)dst,
      0, 0);
#else
  *(u32x4*)dst = *(const u32x4*)src;
#endif
}

__device__ __forceinline__ void wait_cp() {
#if USE_ASYNC_LDS
  __builtin_amdgcn_s_wait_asynccnt(0);
#endif
}

// ---------------------------------------------------------------- convert f32 -> bf16
__global__ void __launch_bounds__(256) k_cvt_bf16(const float* __restrict__ in,
                                                  unsigned short* __restrict__ out,
                                                  int n4) {
  int i = blockIdx.x * blockDim.x + threadIdx.x;
  if (i >= n4) return;
  f32x4 f = *(const f32x4*)(in + (size_t)i * 4);
  u16x4 r;
  r[0] = f2bf(f[0]); r[1] = f2bf(f[1]); r[2] = f2bf(f[2]); r[3] = f2bf(f[3]);
  *(u16x4*)(out + (size_t)i * 4) = r;
}

// ---------------------------------------------------------------- shared GEMM core
// 128-thread block (4 waves); workgroup tile 128x64; wave w owns rows
// [32w, 32w+32) x 64 cols -> acc[2][4] 16x16 f32 tiles, 8 WMMAs per K-step.
// LDS tiles double-buffered; stride 40 (pad) to avoid bank conflicts.
#define SA_STRIDE 40
#define SA_TILE   (128 * SA_STRIDE)
#define SW_TILE   (64 * SA_STRIDE)

__device__ __forceinline__ void stage_tiles(const unsigned short* __restrict__ A,
                                            const unsigned short* __restrict__ W,
                                            int mBlock, int jBlock, int k0,
                                            unsigned short* cA, unsigned short* cW,
                                            int tid) {
#pragma unroll
  for (int rep = 0; rep < 4; ++rep) {      // A: 128 rows x 32 k = 512 x b128
    int idx = tid + rep * 128;
    int row = idx >> 2, qc = idx & 3;
    cp_g2l_b128(A + (size_t)(mBlock + row) * DIM + k0 + qc * 8,
                cA + row * SA_STRIDE + qc * 8);
  }
#pragma unroll
  for (int rep = 0; rep < 2; ++rep) {      // W: 64 rows x 32 k = 256 x b128
    int idx = tid + rep * 128;
    int row = idx >> 2, qc = idx & 3;
    cp_g2l_b128(W + (size_t)(jBlock + row) * DIM + k0 + qc * 8,
                cW + row * SA_STRIDE + qc * 8);
  }
}

__device__ __forceinline__ void gemm_tile_128x64(const unsigned short* __restrict__ A,
                                                 const unsigned short* __restrict__ W,
                                                 int mBlock, int jBlock,
                                                 unsigned short* sA,  // 2*SA_TILE
                                                 unsigned short* sW,  // 2*SW_TILE
                                                 v8f acc[8]) {
  const int tid  = threadIdx.x;
  const int w    = tid >> 5;
  const int lane = tid & 31;
  const int c    = lane & 15;
  const int h    = lane >> 4;
#pragma unroll
  for (int t = 0; t < 8; ++t) acc[t] = v8f_zero();

  stage_tiles(A, W, mBlock, jBlock, 0, sA, sW, tid);

  int buf = 0;
  for (int k0 = 0; k0 < DIM; k0 += 32, buf ^= 1) {
    wait_cp();
    __syncthreads();                 // tiles in buf are complete block-wide
    if (k0 + 32 < DIM)               // issue next K-step into the other buffer
      stage_tiles(A, W, mBlock, jBlock, k0 + 32,
                  sA + (buf ^ 1) * SA_TILE, sW + (buf ^ 1) * SW_TILE, tid);
#if !USE_ASYNC_LDS
    if (k0 + 64 < DIM)
      __builtin_prefetch(A + (size_t)(mBlock + (tid & 127)) * DIM + k0 + 64, 0, 1);
#endif
    const unsigned short* cA = sA + buf * SA_TILE;
    const unsigned short* cW = sW + buf * SW_TILE;

    // A fragments: lane holds row, k = (i/8)*16 + h*8 + i%8
    FragBF a0, a1;
    a0.q[0] = *(const u32x4*)(cA + (w * 32 + c) * SA_STRIDE + h * 8);
    a0.q[1] = *(const u32x4*)(cA + (w * 32 + c) * SA_STRIDE + h * 8 + 16);
    a1.q[0] = *(const u32x4*)(cA + (w * 32 + 16 + c) * SA_STRIDE + h * 8);
    a1.q[1] = *(const u32x4*)(cA + (w * 32 + 16 + c) * SA_STRIDE + h * 8 + 16);
#pragma unroll
    for (int t = 0; t < 4; ++t) {
      // B fragment: lane holds output col (t*16+c), k = (i/8)*16 + h*8 + i%8
      FragBF bf;
      bf.q[0] = *(const u32x4*)(cW + (t * 16 + c) * SA_STRIDE + h * 8);
      bf.q[1] = *(const u32x4*)(cW + (t * 16 + c) * SA_STRIDE + h * 8 + 16);
      acc[t]     = wmma_bf16(a0.v, bf.v, acc[t]);
      acc[4 + t] = wmma_bf16(a1.v, bf.v, acc[4 + t]);
    }
    // no trailing barrier needed: each wave's ds_loads are consumed (waited)
    // before it reaches the next top-of-loop barrier, so WAR on buf is safe.
  }
}

// ---------------------------------------------------------------- QKV GEMM + split
__global__ void __launch_bounds__(128) k_qkv_gemm(const unsigned short* __restrict__ xb,
                                                  const unsigned short* __restrict__ wq,
                                                  const float* __restrict__ qkv_b,
                                                  unsigned short* __restrict__ Q,
                                                  unsigned short* __restrict__ K,
                                                  unsigned short* __restrict__ VT) {
  __shared__ __align__(16) unsigned short sA[2 * SA_TILE];
  __shared__ __align__(16) unsigned short sW[2 * SW_TILE];
  const int mBlock = blockIdx.y * 128;
  const int jBlock = blockIdx.x * 64;
  v8f acc[8];
  gemm_tile_128x64(xb, wq, mBlock, jBlock, sA, sW, acc);

  const int tid = threadIdx.x, w = tid >> 5, lane = tid & 31;
  const int c = lane & 15, h = lane >> 4;
#pragma unroll
  for (int g = 0; g < 2; ++g) {
#pragma unroll
    for (int t = 0; t < 4; ++t) {
      int j = jBlock + t * 16 + c;
      float bias = qkv_b[j];
#pragma unroll
      for (int r = 0; r < 8; ++r) {
        int row = mBlock + w * 32 + g * 16 + r + 8 * h;  // C layout: m = r + 8h
        float val = acc[g * 4 + t][r] + bias;
        int b = row >> 11, n = row & 2047;
        if (j < DIM) {                           // Q, pre-scaled
          int head = j >> 6, d = j & 63;
          Q[(((size_t)(b * HEADS + head)) * SEQ + n) * HD + d] = f2bf(val * ATTN_SCALE);
        } else if (j < 2 * DIM) {                // K
          int jj = j - DIM, head = jj >> 6, d = jj & 63;
          K[(((size_t)(b * HEADS + head)) * SEQ + n) * HD + d] = f2bf(val);
        } else {                                 // V, stored transposed [Hd, N]
          int jj = j - 2 * DIM, head = jj >> 6, d = jj & 63;
          VT[(((size_t)(b * HEADS + head)) * HD + d) * SEQ + n] = f2bf(val);
        }
      }
    }
  }
}

// ---------------------------------------------------------------- flash attention
__global__ void __launch_bounds__(128) k_flash_attn(const unsigned short* __restrict__ Q,
                                                    const unsigned short* __restrict__ K,
                                                    const unsigned short* __restrict__ VT,
                                                    unsigned short* __restrict__ O) {
  __shared__ __align__(16) unsigned short P[4][16 * 40];  // per-wave P staging
  const int tid = threadIdx.x, w = tid >> 5, lane = tid & 31;
  const int c = lane & 15, h = lane >> 4;
  const int bh = blockIdx.z * HEADS + blockIdx.y;
  const unsigned short* Qb = Q + (size_t)bh * SEQ * HD;
  const unsigned short* Kb = K + (size_t)bh * SEQ * HD;
  const unsigned short* Vb = VT + (size_t)bh * HD * SEQ;
  const int qrow = blockIdx.x * 64 + w * 16 + c;

  // Q fragments for d-chunks 0/1 (reused for all key tiles)
  FragBF qf0, qf1;
  qf0.q[0] = *(const u32x4*)(Qb + qrow * HD + h * 8);
  qf0.q[1] = *(const u32x4*)(Qb + qrow * HD + h * 8 + 16);
  qf1.q[0] = *(const u32x4*)(Qb + qrow * HD + 32 + h * 8);
  qf1.q[1] = *(const u32x4*)(Qb + qrow * HD + 32 + h * 8 + 16);

  v8f o[4];
#pragma unroll
  for (int t = 0; t < 4; ++t) o[t] = v8f_zero();
  float rmax[8], rsum[8];
#pragma unroll
  for (int r = 0; r < 8; ++r) { rmax[r] = -1.0e30f; rsum[r] = 0.0f; }
  unsigned short* Pw = P[w];

  for (int kb = 0; kb < SEQ; kb += 32) {
    FragBF kf;
    v8f s0 = v8f_zero(), s1 = v8f_zero();
    // S0: keys kb..kb+15
    kf.q[0] = *(const u32x4*)(Kb + (kb + c) * HD + h * 8);
    kf.q[1] = *(const u32x4*)(Kb + (kb + c) * HD + h * 8 + 16);
    s0 = wmma_bf16(qf0.v, kf.v, s0);
    kf.q[0] = *(const u32x4*)(Kb + (kb + c) * HD + 32 + h * 8);
    kf.q[1] = *(const u32x4*)(Kb + (kb + c) * HD + 32 + h * 8 + 16);
    s0 = wmma_bf16(qf1.v, kf.v, s0);
    // S1: keys kb+16..kb+31
    kf.q[0] = *(const u32x4*)(Kb + (kb + 16 + c) * HD + h * 8);
    kf.q[1] = *(const u32x4*)(Kb + (kb + 16 + c) * HD + h * 8 + 16);
    s1 = wmma_bf16(qf0.v, kf.v, s1);
    kf.q[0] = *(const u32x4*)(Kb + (kb + 16 + c) * HD + 32 + h * 8);
    kf.q[1] = *(const u32x4*)(Kb + (kb + 16 + c) * HD + 32 + h * 8 + 16);
    s1 = wmma_bf16(qf1.v, kf.v, s1);

    // online softmax over this 32-key slab (row m = r + 8h lives in 16 lanes)
#pragma unroll
    for (int r = 0; r < 8; ++r) {
      float v0 = s0[r], v1 = s1[r];
      float mx = fmaxf(v0, v1);
      mx = fmaxf(mx, __shfl_xor(mx, 1, 32));
      mx = fmaxf(mx, __shfl_xor(mx, 2, 32));
      mx = fmaxf(mx, __shfl_xor(mx, 4, 32));
      mx = fmaxf(mx, __shfl_xor(mx, 8, 32));
      float mnew = fmaxf(rmax[r], mx);
      float corr = __expf(rmax[r] - mnew);
      rmax[r] = mnew;
      float p0 = __expf(v0 - mnew);
      float p1 = __expf(v1 - mnew);
      float ps = p0 + p1;
      ps += __shfl_xor(ps, 1, 32);
      ps += __shfl_xor(ps, 2, 32);
      ps += __shfl_xor(ps, 4, 32);
      ps += __shfl_xor(ps, 8, 32);
      rsum[r] = rsum[r] * corr + ps;
      o[0][r] *= corr; o[1][r] *= corr; o[2][r] *= corr; o[3][r] *= corr;
      int m = r + 8 * h;
      Pw[m * 40 + c]      = f2bf(p0);   // C-layout -> LDS (A-layout transpose)
      Pw[m * 40 + 16 + c] = f2bf(p1);
    }
    // P as A-fragment (16 rows x 32 keys); same-wave LDS ops are in-order
    FragBF pf;
    pf.q[0] = *(const u32x4*)(Pw + c * 40 + h * 8);
    pf.q[1] = *(const u32x4*)(Pw + c * 40 + h * 8 + 16);
#pragma unroll
    for (int t = 0; t < 4; ++t) {
      FragBF vf;  // V^T: lane holds out-col d = t*16+c, keys (i/8)*16 + h*8 + i%8
      vf.q[0] = *(const u32x4*)(Vb + (t * 16 + c) * SEQ + kb + h * 8);
      vf.q[1] = *(const u32x4*)(Vb + (t * 16 + c) * SEQ + kb + h * 8 + 16);
      o[t] = wmma_bf16(pf.v, vf.v, o[t]);
    }
  }

  float rinv[8];
#pragma unroll
  for (int r = 0; r < 8; ++r) rinv[r] = 1.0f / rsum[r];
#pragma unroll
  for (int t = 0; t < 4; ++t) {
#pragma unroll
    for (int r = 0; r < 8; ++r) {
      int m = r + 8 * h;
      int n = blockIdx.x * 64 + w * 16 + m;
      size_t row = (size_t)blockIdx.z * SEQ + n;
      O[row * DIM + blockIdx.y * HD + t * 16 + c] = f2bf(o[t][r] * rinv[r]);
    }
  }
}

// ---------------------------------------------------------------- output projection
__global__ void __launch_bounds__(128) k_proj_gemm(const unsigned short* __restrict__ Ob,
                                                   const unsigned short* __restrict__ wp,
                                                   const float* __restrict__ proj_b,
                                                   float* __restrict__ out) {
  __shared__ __align__(16) unsigned short sA[2 * SA_TILE];
  __shared__ __align__(16) unsigned short sW[2 * SW_TILE];
  const int mBlock = blockIdx.y * 128;
  const int jBlock = blockIdx.x * 64;
  v8f acc[8];
  gemm_tile_128x64(Ob, wp, mBlock, jBlock, sA, sW, acc);

  const int tid = threadIdx.x, w = tid >> 5, lane = tid & 31;
  const int c = lane & 15, h = lane >> 4;
#pragma unroll
  for (int g = 0; g < 2; ++g) {
#pragma unroll
    for (int t = 0; t < 4; ++t) {
      int j = jBlock + t * 16 + c;
      float bias = proj_b[j];
#pragma unroll
      for (int r = 0; r < 8; ++r) {
        int row = mBlock + w * 32 + g * 16 + r + 8 * h;
        out[(size_t)row * DIM + j] = acc[g * 4 + t][r] + bias;
      }
    }
  }
}

// ---------------------------------------------------------------- launch
extern "C" void kernel_launch(void* const* d_in, const int* in_sizes, int n_in,
                              void* d_out, int out_size, void* d_ws, size_t ws_size,
                              hipStream_t stream) {
  (void)in_sizes; (void)n_in; (void)out_size; (void)ws_size;
  const float* x      = (const float*)d_in[0];
  const float* qkv_w  = (const float*)d_in[1];
  const float* qkv_b  = (const float*)d_in[2];
  const float* proj_w = (const float*)d_in[3];
  const float* proj_b = (const float*)d_in[4];
  float* out = (float*)d_out;

  const size_t SZ_X  = (size_t)MTOT * DIM;   // 6,291,456
  const size_t SZ_WQ = (size_t)NQKV * DIM;   // 1,769,472
  const size_t SZ_WP = (size_t)DIM * DIM;    //   589,824

  char* ws = (char*)d_ws;
  unsigned short* xb  = (unsigned short*)ws;  ws += SZ_X * 2;
  unsigned short* wqb = (unsigned short*)ws;  ws += SZ_WQ * 2;
  unsigned short* wpb = (unsigned short*)ws;  ws += SZ_WP * 2;
  unsigned short* Qb  = (unsigned short*)ws;  ws += SZ_X * 2;
  unsigned short* Kb  = (unsigned short*)ws;  ws += SZ_X * 2;
  unsigned short* VTb = (unsigned short*)ws;  ws += SZ_X * 2;
  unsigned short* Ob  = (unsigned short*)ws;  ws += SZ_X * 2;

  k_cvt_bf16<<<(int)(SZ_X / 4 / 256),  256, 0, stream>>>(x,      xb,  (int)(SZ_X / 4));
  k_cvt_bf16<<<(int)(SZ_WQ / 4 / 256), 256, 0, stream>>>(qkv_w,  wqb, (int)(SZ_WQ / 4));
  k_cvt_bf16<<<(int)(SZ_WP / 4 / 256), 256, 0, stream>>>(proj_w, wpb, (int)(SZ_WP / 4));

  k_qkv_gemm<<<dim3(NQKV / 64, MTOT / 128), 128, 0, stream>>>(xb, wqb, qkv_b, Qb, Kb, VTb);
  k_flash_attn<<<dim3(SEQ / 64, HEADS, BATCH), 128, 0, stream>>>(Qb, Kb, VTb, Ob);
  k_proj_gemm<<<dim3(DIM / 64, MTOT / 128), 128, 0, stream>>>(Ob, wpb, proj_b, out);
}
